// MultiWindowHeadCrossAttention_35132832481299
// MI455X (gfx1250) — compile-verified, matrix operand order checked by reading, not run on hardware
//
#include <hip/hip_runtime.h>
#include <hip/hip_bf16.h>
#include <math.h>

// ---------------------------------------------------------------------------
// Types for CDNA5 WMMA (wave32, 16x16x32 bf16 -> f32)
// ---------------------------------------------------------------------------
typedef __attribute__((ext_vector_type(16))) __bf16 v16bf;
typedef __attribute__((ext_vector_type(8)))  __bf16 v8bf;
typedef __attribute__((ext_vector_type(8)))  float  v8f;

__device__ __forceinline__ v8f wmma_bf16(v16bf a, v16bf b, v8f c) {
    // D = A(16x32) * B(32x16) + C, bf16 inputs, f32 accum
    return __builtin_amdgcn_wmma_f32_16x16x32_bf16(
        /*neg_a=*/false, a, /*neg_b=*/false, b,
        /*c_mod=*/(short)0, c, /*reuse_a=*/false, /*reuse_b=*/false);
}

// Load 16 bf16 A/B-fragment elements for one K-chunk from a row-major row.
// p points at (row_base + chunk + off); reads p[0..7] and p[16..23].
__device__ __forceinline__ v16bf ldA_bf16(const __bf16* __restrict__ p) {
    v8bf lo = *(const v8bf*)(p);
    v8bf hi = *(const v8bf*)(p + 16);
    v16bf r;
#pragma unroll
    for (int i = 0; i < 8; ++i) { r[i] = lo[i]; r[8 + i] = hi[i]; }
    return r;
}

// ---------------------------------------------------------------------------
// Elementwise f32 -> bf16 conversion (8 elements / thread, b128 in, b128 out)
// ---------------------------------------------------------------------------
__global__ __launch_bounds__(256) void cvt_f32_bf16(
    const float* __restrict__ x, __bf16* __restrict__ y, int n8) {
    const int i = blockIdx.x * blockDim.x + threadIdx.x;
    if (i >= n8) return;
    const float4* p = (const float4*)x + 2 * (size_t)i;
    const float4 a = p[0];
    const float4 b = p[1];
    v8bf r;
    r[0] = (__bf16)a.x; r[1] = (__bf16)a.y; r[2] = (__bf16)a.z; r[3] = (__bf16)a.w;
    r[4] = (__bf16)b.x; r[5] = (__bf16)b.y; r[6] = (__bf16)b.z; r[7] = (__bf16)b.w;
    ((v8bf*)y)[i] = r;
}

// ---------------------------------------------------------------------------
// GEMM: Y[M,N] = X[M,K] @ W[N,K]^T + bias, all-bf16 operands (pre-converted).
// NW=2 computes two outputs sharing the A operand (K and V projections).
// One wave -> 16x64 output tile per output; 4 waves/block -> 16x256.
// ---------------------------------------------------------------------------
template <int NW, bool OUT_F32>
__global__ __launch_bounds__(128) void gemm_xwT_bias(
    const __bf16* __restrict__ X,
    const __bf16* __restrict__ W0, const float* __restrict__ b0, void* __restrict__ Y0,
    const __bf16* __restrict__ W1, const float* __restrict__ b1, void* __restrict__ Y1,
    int M, int K, int N) {
    const int lane = threadIdx.x & 31;
    const int wv   = threadIdx.x >> 5;
    const int lr   = lane & 15;
    const int hf   = lane >> 4;
    const int off  = hf * 8;
    const int M0   = blockIdx.x * 16;
    const int N0   = blockIdx.y * 256 + wv * 64;

    const size_t rowA = (size_t)(M0 + lr) * (size_t)K;
    v8f acc0[4] = {};
    v8f acc1[4] = {};

    for (int c = 0; c < K; c += 32) {
        const __bf16* ap = X + rowA + c + off;
        __builtin_prefetch(ap + 64, 0, 0);      // global_prefetch_b8: next chunks
        v16bf a = ldA_bf16(ap);
#pragma unroll
        for (int t = 0; t < 4; ++t) {
            const size_t wrow = (size_t)(N0 + t * 16 + lr) * (size_t)K + c + off;
            v16bf bw0 = ldA_bf16(W0 + wrow);
            acc0[t] = wmma_bf16(a, bw0, acc0[t]);
            if constexpr (NW == 2) {
                v16bf bw1 = ldA_bf16(W1 + wrow);
                acc1[t] = wmma_bf16(a, bw1, acc1[t]);
            }
        }
    }

#pragma unroll
    for (int t = 0; t < 4; ++t) {
        const int col = N0 + t * 16 + lr;
        const float bias0 = b0[col];
        const float bias1 = (NW == 2) ? b1[col] : 0.0f;
#pragma unroll
        for (int j = 0; j < 8; ++j) {
            const int row = M0 + j + 8 * hf;
            const size_t idx = (size_t)row * N + col;
            const float v0 = acc0[t][j] + bias0;
            if constexpr (OUT_F32)
                ((float*)Y0)[idx] = v0;
            else
                ((__bf16*)Y0)[idx] = (__bf16)v0;
            if constexpr (NW == 2)
                ((__bf16*)Y1)[idx] = (__bf16)(acc1[t][j] + bias1);
        }
    }
}

// ---------------------------------------------------------------------------
// Sliding-window (W=16) attention, one wave per 16-query tile.
// Scores vs two 16-key tiles (band), masked softmax, P@V via WMMA.
// Reproduces the reference's zero-padded keys and w<=i causal mask exactly.
// ---------------------------------------------------------------------------
__global__ __launch_bounds__(128) void attn_window16(
    const __bf16* __restrict__ Q, const __bf16* __restrict__ Kb,
    const __bf16* __restrict__ Vb, __bf16* __restrict__ O,
    const int* __restrict__ causal_flag, int S, int Hc) {
    __shared__ float  pbuf[4][16 * 36];       // P (16x32), padded stride 36
    __shared__ __bf16 vtile[4][32 * 72];      // V band (32x64), padded stride 72

    const int w    = threadIdx.x >> 5;
    const int lane = threadIdx.x & 31;
    const int lr   = lane & 15;
    const int hf   = lane >> 4;
    const int off  = hf * 8;
    const int bh   = blockIdx.x;
    const int b    = bh / Hc;
    const int h    = bh % Hc;
    const int s0   = (blockIdx.y * 4 + w) * 16;
    const int D    = Hc * 64;
    const bool causal = (causal_flag[0] != 0);

    auto rowp = [&](int s) -> size_t { return (size_t)(b * S + s) * (size_t)D + (size_t)h * 64; };

    // ---- Q fragments (A operand, K-dim 64 = two chunks of 32)
    const __bf16* qrow = Q + rowp(s0 + lr) + off;
    v16bf qa0 = ldA_bf16(qrow);
    v16bf qa1 = ldA_bf16(qrow + 32);

    // ---- K fragments (B operand). Tile0 keys j=s0-16+n (zero rows when j<0).
    v16bf kb00 = {}, kb01 = {};
    const int key0 = s0 - 16 + lr;
    if (key0 >= 0) {
        const __bf16* kr = Kb + rowp(key0) + off;
        kb00 = ldA_bf16(kr);
        kb01 = ldA_bf16(kr + 32);
    }
    const __bf16* kr1 = Kb + rowp(s0 + lr) + off;
    v16bf kb10 = ldA_bf16(kr1);
    v16bf kb11 = ldA_bf16(kr1 + 32);

    // ---- scores: S0 = Q @ K0^T, S1 = Q @ K1^T   (16x16 f32 each)
    v8f sA = {}, sB = {};
    sA = wmma_bf16(qa0, kb00, sA);
    sA = wmma_bf16(qa1, kb01, sA);
    sB = wmma_bf16(qa0, kb10, sB);
    sB = wmma_bf16(qa1, kb11, sB);

    // ---- stage V band into LDS (zero rows for j<0, matching the zero pad)
    {
        const int vkey = s0 - 16 + lane;
        uint4* vd = (uint4*)&vtile[w][lane * 72];
        if (vkey >= 0) {
            const uint4* vs = (const uint4*)(Vb + rowp(vkey));
#pragma unroll
            for (int i = 0; i < 8; ++i) vd[i] = vs[i];
        } else {
            const uint4 z = {0u, 0u, 0u, 0u};
#pragma unroll
            for (int i = 0; i < 8; ++i) vd[i] = z;
        }
    }

    // ---- masked softmax across the 32-wide band (rows split across lanes)
    const float scale = 0.125f;   // 1/sqrt(64)
    const int n = lr;
#pragma unroll
    for (int j = 0; j < 8; ++j) {
        const int m = j + 8 * hf;
        const bool v0 = (n > m)  && (!causal || (n <= s0 + 2 * m + 1));
        const bool v1 = (n <= m) && (!causal || (n + 15 <= s0 + 2 * m));
        float x0 = v0 ? sA[j] * scale : -__builtin_inff();
        float x1 = v1 ? sB[j] * scale : -__builtin_inff();
        float mx = fmaxf(x0, x1);
#pragma unroll
        for (int d = 1; d < 16; d <<= 1) mx = fmaxf(mx, __shfl_xor(mx, d, 32));
        const float e0 = __expf(x0 - mx);
        const float e1 = __expf(x1 - mx);
        float sm = e0 + e1;
#pragma unroll
        for (int d = 1; d < 16; d <<= 1) sm += __shfl_xor(sm, d, 32);
        const float inv = 1.0f / sm;
        pbuf[w][m * 36 + n]      = e0 * inv;
        pbuf[w][m * 36 + 16 + n] = e1 * inv;
    }
    __syncthreads();

    // ---- P (16x32) as A operand from LDS (C-layout -> A-layout transpose)
    const float* pr = &pbuf[w][lr * 36 + off];
    float4 p0 = *(const float4*)(pr);
    float4 p1 = *(const float4*)(pr + 4);
    float4 p2 = *(const float4*)(pr + 16);
    float4 p3 = *(const float4*)(pr + 20);
    v16bf pa;
    pa[0]  = (__bf16)p0.x; pa[1]  = (__bf16)p0.y; pa[2]  = (__bf16)p0.z; pa[3]  = (__bf16)p0.w;
    pa[4]  = (__bf16)p1.x; pa[5]  = (__bf16)p1.y; pa[6]  = (__bf16)p1.z; pa[7]  = (__bf16)p1.w;
    pa[8]  = (__bf16)p2.x; pa[9]  = (__bf16)p2.y; pa[10] = (__bf16)p2.z; pa[11] = (__bf16)p2.w;
    pa[12] = (__bf16)p3.x; pa[13] = (__bf16)p3.y; pa[14] = (__bf16)p3.z; pa[15] = (__bf16)p3.w;

    // ---- out = P @ Vband (16x32 @ 32x64), 4 N-tiles of 16
    v8f o[4] = {};
#pragma unroll
    for (int t = 0; t < 4; ++t) {
        v16bf vb;
#pragma unroll
        for (int i = 0; i < 8; ++i) {
            vb[i]     = vtile[w][(off + i)      * 72 + t * 16 + lr];
            vb[8 + i] = vtile[w][(off + 16 + i) * 72 + t * 16 + lr];
        }
        o[t] = wmma_bf16(pa, vb, o[t]);
    }

    // ---- store attention output (bf16, same (B*S, D) layout as projections)
#pragma unroll
    for (int t = 0; t < 4; ++t) {
#pragma unroll
        for (int j = 0; j < 8; ++j) {
            const int m = j + 8 * hf;
            O[rowp(s0 + m) + t * 16 + lr] = (__bf16)o[t][j];
        }
    }
}

// ---------------------------------------------------------------------------
// Launch: pre-convert activations+weights to bf16, Q projection, fused K+V
// projection (shared A), fused window attention, output projection (f32 out).
// ---------------------------------------------------------------------------
extern "C" void kernel_launch(void* const* d_in, const int* in_sizes, int n_in,
                              void* d_out, int out_size, void* d_ws, size_t ws_size,
                              hipStream_t stream) {
    (void)in_sizes; (void)n_in; (void)out_size; (void)ws_size;
    const float* query = (const float*)d_in[0];
    const float* kv    = (const float*)d_in[1];
    const float* Wq    = (const float*)d_in[2];
    const float* bq    = (const float*)d_in[3];
    const float* Wk    = (const float*)d_in[4];
    const float* bk    = (const float*)d_in[5];
    const float* Wv    = (const float*)d_in[6];
    const float* bv    = (const float*)d_in[7];
    const float* Wo    = (const float*)d_in[8];
    const float* bo    = (const float*)d_in[9];
    const int* is_causal = (const int*)d_in[10];
    float* out = (float*)d_out;

    const int B = 2, S = 4096, D = 1024, H = 16;
    const int M = B * S;                        // 8192
    const size_t elems = (size_t)M * D;         // 8.39M
    const size_t welems = (size_t)D * D;        // 1.05M

    __bf16* p   = (__bf16*)d_ws;                // bf16 workspace layout
    __bf16* Xq  = p; p += elems;                // query  (bf16)
    __bf16* Xkv = p; p += elems;                // kv     (bf16)
    __bf16* Wqb = p; p += welems;
    __bf16* Wkb = p; p += welems;
    __bf16* Wvb = p; p += welems;
    __bf16* Wob = p; p += welems;
    __bf16* Qb  = p; p += elems;
    __bf16* Kb  = p; p += elems;
    __bf16* Vb  = p; p += elems;
    __bf16* Ob  = p; p += elems;                // total ~109 MB

    // ---- stage 0: f32 -> bf16 conversions (bandwidth-trivial)
    {
        const int tb = 256;
        const int nx = (int)(elems / 8), nw = (int)(welems / 8);
        hipLaunchKernelGGL(cvt_f32_bf16, dim3((nx + tb - 1) / tb), dim3(tb), 0, stream, query, Xq,  nx);
        hipLaunchKernelGGL(cvt_f32_bf16, dim3((nx + tb - 1) / tb), dim3(tb), 0, stream, kv,    Xkv, nx);
        hipLaunchKernelGGL(cvt_f32_bf16, dim3((nw + tb - 1) / tb), dim3(tb), 0, stream, Wq,    Wqb, nw);
        hipLaunchKernelGGL(cvt_f32_bf16, dim3((nw + tb - 1) / tb), dim3(tb), 0, stream, Wk,    Wkb, nw);
        hipLaunchKernelGGL(cvt_f32_bf16, dim3((nw + tb - 1) / tb), dim3(tb), 0, stream, Wv,    Wvb, nw);
        hipLaunchKernelGGL(cvt_f32_bf16, dim3((nw + tb - 1) / tb), dim3(tb), 0, stream, Wo,    Wob, nw);
    }

    dim3 blk(128);
    dim3 gg(M / 16, D / 256);                   // (512, 4)

    // ---- stage 1: Q projection
    hipLaunchKernelGGL((gemm_xwT_bias<1, false>), gg, blk, 0, stream,
                       Xq, Wqb, bq, (void*)Qb, nullptr, nullptr, nullptr, M, D, D);
    // ---- stage 2: K and V projections sharing the kv A operand
    hipLaunchKernelGGL((gemm_xwT_bias<2, false>), gg, blk, 0, stream,
                       Xkv, Wkb, bk, (void*)Kb, Wvb, bv, (void*)Vb, M, D, D);

    // ---- stage 3: banded attention (W=16), one wave per 16-query tile
    dim3 ga(B * H, S / 64);                     // (32, 64)
    hipLaunchKernelGGL(attn_window16, ga, blk, 0, stream,
                       Qb, Kb, Vb, Ob, is_causal, S, H);

    // ---- stage 4: output projection, f32 result
    hipLaunchKernelGGL((gemm_xwT_bias<1, true>), gg, blk, 0, stream,
                       Ob, Wob, bo, (void*)out, nullptr, nullptr, nullptr, M, D, D);
}
